// CluTSPSolver_75136157876542
// MI455X (gfx1250) — matrix operand order repaired and checked by reading, not compiled, not run on hardware
//
#include <hip/hip_runtime.h>
#include <hip/hip_bf16.h>
#include <math.h>

#define Bsz 128
#define Nn  2000
#define Cc  100
#define Dd  128
#define Hh  8
#define DKk 16
#define MT  7          // ceil(100/16) M tiles, padded rows -> 112
#define MPAD 112
#define SEG 10         // N-split for the streaming mean (2000 = 10 * 200)
#define ROWS_PER_SEG 200
#define NEGV (-1000000000.0f)

typedef __attribute__((ext_vector_type(16))) __bf16 v16bf;
typedef __attribute__((ext_vector_type(8)))  float  v8f;
typedef __attribute__((ext_vector_type(4)))  float  f4;

__device__ __forceinline__ __bf16 f2bf(float f) {
    unsigned int u = __builtin_bit_cast(unsigned int, f);
    unsigned int r = (u + 0x7FFFu + ((u >> 16) & 1u)) >> 16;   // round-to-nearest-even
    unsigned short s = (unsigned short)r;
    return __builtin_bit_cast(__bf16, s);
}

// ---------------------------------------------------------------------------
// K1: pack cluster embeddings into WMMA bf16 A-layout (16x32 tiles).
// flat index = ((b*7 + mtile)*4 + kk)*512 + lane*16 + j
// A layout (ISA 7.12.2, 16-bit A 16x32): lane<16: row=lane, K = {j<8 ? j : j+8};
// lane>=16: row=lane-16, K += 8. Tile K base = kk*32.
// ---------------------------------------------------------------------------
__global__ void k_packA(const float* __restrict__ clu, __bf16* __restrict__ outp) {
    int idx   = blockIdx.x * 256 + threadIdx.x;      // < 128*7*4*512 = 1,835,008
    int j     = idx & 15;
    int lane  = (idx >> 4) & 31;
    int kk    = (idx >> 9) & 3;
    int mtile = (idx >> 11) % MT;
    int b     = idx / (MT * 2048);
    int row   = mtile * 16 + (lane & 15);
    int k     = kk * 32 + ((j < 8) ? j : j + 8) + ((lane < 16) ? 0 : 8);
    float v   = (row < Cc) ? clu[((size_t)b * Cc + row) * Dd + k] : 0.0f;
    outp[idx] = f2bf(v);
}

// ---------------------------------------------------------------------------
// K2: fuse Wkk = Wk@Wmk, Wvv = Wv@Wmv (once, not per batch) and pack them +
// Wks into WMMA bf16 B-layout (32x16): lane<16: col=lane, K={j<8 ? j : j+8}; lane>=16: K += 8.
// flat = (ntile*4 + kk)*512 + lane*16 + j
// ---------------------------------------------------------------------------
__global__ void k_packW(const float* __restrict__ Wk,  const float* __restrict__ Wmk,
                        const float* __restrict__ Wv,  const float* __restrict__ Wmv,
                        const float* __restrict__ Wks,
                        __bf16* __restrict__ pKK, __bf16* __restrict__ pVV,
                        __bf16* __restrict__ pKS) {
    int t    = blockIdx.x * 256 + threadIdx.x;       // < 3*16384
    int mat  = t >> 14;
    int idx  = t & 16383;
    int j    = idx & 15;
    int lane = (idx >> 4) & 31;
    int kk   = (idx >> 9) & 3;
    int ntile= idx >> 11;                            // 0..7
    int col  = ntile * 16 + (lane & 15);
    int k    = kk * 32 + ((j < 8) ? j : j + 8) + ((lane < 16) ? 0 : 8);
    float v;
    if (mat == 2) {
        v = Wks[k * Dd + col];
    } else {
        const float* A  = (mat == 0) ? Wk  : Wv;
        const float* Bm = (mat == 0) ? Wmk : Wmv;
        float s = 0.0f;
        for (int q = 0; q < Dd; ++q) s += A[k * Dd + q] * Bm[q * Dd + col];
        v = s;
    }
    __bf16* dst = (mat == 0) ? pKK : (mat == 1) ? pVV : pKS;
    dst[idx] = f2bf(v);
}

// K3: Wqq = Wq @ Wmq  ([384,128]@[128,128], f32, once)
__global__ void k_wqq(const float* __restrict__ Wq, const float* __restrict__ Wmq,
                      float* __restrict__ Wqq) {
    int t = blockIdx.x * 256 + threadIdx.x;          // < 384*128
    int d = t & 127;
    int r = t >> 7;
    float s = 0.0f;
    for (int q = 0; q < Dd; ++q) s += Wq[r * Dd + q] * Wmq[q * Dd + d];
    Wqq[t] = s;
}

// ---------------------------------------------------------------------------
// K4a: streaming pass over node_embeddings (131 MB -> the HBM roofline term).
// grid = (SEG, B) = 1280 blocks for occupancy; each wave streams one full
// 512B row per global_load_b128 (float4 per lane). Produces per-segment
// partial sums for BOTH masks (deterministic: no float atomics).
// ---------------------------------------------------------------------------
__global__ void k_means_part(const float* __restrict__ node, const int* __restrict__ mask,
                             const int* __restrict__ cmask,
                             float* __restrict__ pum, float* __restrict__ puc) {
    __shared__ f4 s0[256], s1[256];
    int b   = blockIdx.y;
    int seg = blockIdx.x;
    int tid = threadIdx.x;
    int rowoff = tid >> 5;          // wave id 0..7 -> row within group of 8
    int d0     = (tid & 31) * 4;    // contiguous float4 per lane
    const float* np = node + ((size_t)b * Nn + (size_t)seg * ROWS_PER_SEG) * Dd;
    const int*   mp = mask  + (size_t)b * Nn + seg * ROWS_PER_SEG;
    const int*   cp = cmask + (size_t)b * Nn + seg * ROWS_PER_SEG;
    f4 a0 = {0.f, 0.f, 0.f, 0.f};
    f4 a1 = {0.f, 0.f, 0.f, 0.f};
    for (int it = 0; it < ROWS_PER_SEG / 8; ++it) {      // 25 iterations
        int n = it * 8 + rowoff;
        f4 x = *(const f4*)(np + (size_t)n * Dd + d0);
        int m  = mp[n];
        int mc = m | cp[n];
        if (!m)  a0 += x;
        if (!mc) a1 += x;
    }
    s0[tid] = a0; s1[tid] = a1;
    __syncthreads();
    if (tid < 32) {                                       // fixed-order combine
        f4 t0 = s0[tid], t1 = s1[tid];
        for (int w = 1; w < 8; ++w) { t0 += s0[tid + w * 32]; t1 += s1[tid + w * 32]; }
        f4* pu = (f4*)(pum + ((size_t)b * SEG + seg) * Dd);
        f4* pc = (f4*)(puc + ((size_t)b * SEG + seg) * Dd);
        pu[tid] = t0;
        pc[tid] = t1;
    }
}

// K4b: fixed-order reduction of SEG partials -> um, uc (deterministic)
__global__ void k_means_reduce(const float* __restrict__ pum, const float* __restrict__ puc,
                               float* __restrict__ um, float* __restrict__ uc) {
    int b = blockIdx.x, d = threadIdx.x;                  // 128 threads
    float s0 = 0.0f, s1 = 0.0f;
    for (int s = 0; s < SEG; ++s) {
        s0 += pum[((size_t)b * SEG + s) * Dd + d];
        s1 += puc[((size_t)b * SEG + s) * Dd + d];
    }
    um[b * Dd + d] = s0 * (1.0f / (float)Nn);
    uc[b * Dd + d] = s1 * (1.0f / (float)Nn);
}

// ---------------------------------------------------------------------------
// K5: WMMA GEMMs. kh = E@(Wk@Wmk), vh = E@(Wv@Wmv), pk = E@Wks, all
// [100x128]@[128x128] per batch, bf16 inputs, f32 accumulate.
// grid = (7 mtiles, 128 batches, 3 matrices); 8 waves/block = 8 N-tiles.
// ---------------------------------------------------------------------------
__global__ void k_gemm(const __bf16* __restrict__ cluA,
                       const __bf16* __restrict__ w0, const __bf16* __restrict__ w1,
                       const __bf16* __restrict__ w2,
                       float* __restrict__ o0, float* __restrict__ o1,
                       float* __restrict__ o2) {
    int mtile = blockIdx.x;              // 0..6
    int b     = blockIdx.y;              // 0..127
    int mat   = blockIdx.z;              // 0..2
    int wave  = threadIdx.x >> 5;        // = ntile 0..7
    int lane  = threadIdx.x & 31;
    const __bf16* Bw  = (mat == 0) ? w0 : (mat == 1) ? w1 : w2;
    float*        Out = ((mat == 0) ? o0 : (mat == 1) ? o1 : o2) + (size_t)b * MPAD * Dd;
    const __bf16* Ab = cluA + ((size_t)(b * MT + mtile) * 4) * 512 + lane * 16;
    const __bf16* Bb = Bw   + ((size_t)wave * 4) * 512 + lane * 16;
    v8f acc = {};
#pragma unroll
    for (int kk = 0; kk < 4; ++kk) {
        v16bf a  = *(const v16bf*)(Ab + kk * 512);   // 32B contiguous per lane
        v16bf bb = *(const v16bf*)(Bb + kk * 512);
        acc = __builtin_amdgcn_wmma_f32_16x16x32_bf16(
            false, a, false, bb, (short)0, acc, false, false);
    }
    // C/D layout: VGPR r -> row = r + (lane<16 ? 0 : 8), col = lane&15
    int row0 = mtile * 16 + ((lane < 16) ? 0 : 8);
    int col  = wave * 16 + (lane & 15);
#pragma unroll
    for (int r = 0; r < 8; ++r)
        Out[(size_t)(row0 + r) * Dd + col] = acc[r];
}

// K6: qh = [um | current | depot] @ Wqq   (per batch 384x128 matvec)
__global__ void k_qpath(const float* __restrict__ um, const float* __restrict__ cur,
                        const float* __restrict__ depot, const float* __restrict__ Wqq,
                        float* __restrict__ qh) {
    __shared__ float ctx[3 * Dd];
    int b = blockIdx.x, t = threadIdx.x;             // 128 threads
    ctx[t]       = um[b * Dd + t];
    ctx[128 + t] = cur[b * Dd + t];
    ctx[256 + t] = depot[b * Dd + t];
    __syncthreads();
    float s = 0.0f;
    for (int j = 0; j < 3 * Dd; ++j) s += ctx[j] * Wqq[j * Dd + t];
    qh[b * Dd + t] = s;
}

// ---------------------------------------------------------------------------
// K7: per-batch epilogue: masked MHA over clusters, Wmo projection, tanh
// logits, log-softmax, argmax, gather + output packing.
// d_out layout (flat, return order): aug[128*512] | guid_emb[128*128] |
// guidance[128] (int32) | clu_prob[128*100]
// ---------------------------------------------------------------------------
__global__ void k_attn(const float* __restrict__ qh, const float* __restrict__ kh,
                       const float* __restrict__ vh, const float* __restrict__ pkm,
                       const float* __restrict__ clu, const float* __restrict__ uc,
                       const float* __restrict__ cur, const float* __restrict__ depot,
                       const int* __restrict__ isnew, const int* __restrict__ visited,
                       const float* __restrict__ Wmo, float* __restrict__ out) {
    __shared__ float sc[Hh * Cc];    // scores -> attn
    __shared__ float gl[Dd];         // glimpse (h-major H*DK)
    __shared__ float go[Dd];         // glimpse @ Wmo
    __shared__ float lg[Cc];         // logits
    __shared__ int   vcm[Cc];
    __shared__ int   s_g;
    __shared__ float s_lse;
    int b = blockIdx.x, t = threadIdx.x;             // 128 threads
    int inew = isnew[b] != 0;
    if (t < Cc) vcm[t] = visited[b * Cc + t] != 0;
    __syncthreads();
    if (t == 0) {
        int all_vis = 1;
        for (int c = 1; c < Cc; ++c) all_vis &= vcm[c];
        vcm[0] = inew ? !all_vis : vcm[0];
    }
    __syncthreads();
    const float* khb = kh  + (size_t)b * MPAD * Dd;
    const float* vhb = vh  + (size_t)b * MPAD * Dd;
    const float* pkb = pkm + (size_t)b * MPAD * Dd;
    const float* qb  = qh + b * Dd;
    // scores[h][c] = <qh[h], kh[c,h]> / sqrt(16), masked
    for (int idx = t; idx < Hh * Cc; idx += 128) {
        int h = idx / Cc, c = idx % Cc;
        float s = 0.0f;
        for (int k = 0; k < DKk; ++k) s += qb[h * DKk + k] * khb[c * Dd + h * DKk + k];
        sc[idx] = vcm[c] ? NEGV : s * 0.25f;
    }
    __syncthreads();
    if (t < Hh) {                                    // per-head softmax over C
        float mx = sc[t * Cc];
        for (int c = 1; c < Cc; ++c) mx = fmaxf(mx, sc[t * Cc + c]);
        float sm = 0.0f;
        for (int c = 0; c < Cc; ++c) { float e = expf(sc[t * Cc + c] - mx); sc[t * Cc + c] = e; sm += e; }
        float inv = 1.0f / sm;
        for (int c = 0; c < Cc; ++c) sc[t * Cc + c] *= inv;
    }
    __syncthreads();
    {                                                // glimpse[h*16+dk]
        int h = t >> 4;
        float a = 0.0f;
        for (int c = 0; c < Cc; ++c) a += sc[h * Cc + c] * vhb[c * Dd + t];
        gl[t] = a;
    }
    __syncthreads();
    {                                                // @ Wmo
        float a = 0.0f;
        for (int j = 0; j < Dd; ++j) a += gl[j] * Wmo[j * Dd + t];
        go[t] = a;
    }
    __syncthreads();
    if (t < Cc) {                                    // logits
        float s = 0.0f;
        for (int d = 0; d < Dd; ++d) s += go[d] * pkb[t * Dd + d];
        s *= 0.08838834764831845f;                   // 1/sqrt(128)
        s = tanhf(s) * 10.0f;
        lg[t] = vcm[t] ? NEGV : s;
    }
    __syncthreads();
    if (t == 0) {                                    // argmax + logsumexp
        float mx = lg[0]; int g = 0;
        for (int c = 1; c < Cc; ++c) if (lg[c] > mx) { mx = lg[c]; g = c; }
        float sm = 0.0f;
        for (int c = 0; c < Cc; ++c) sm += expf(lg[c] - mx);
        s_lse = mx + logf(sm);
        s_g = g;
    }
    __syncthreads();
    int   g   = s_g;
    float fin = inew ? 1.0f : 0.0f;
    if (t < Cc) out[82048 + b * Cc + t] = inew ? (lg[t] - s_lse) : 0.0f;   // clu_prob
    if (t == 0) ((int*)out)[81920 + b] = inew ? g : 0;                     // guidance
    float ge = clu[((size_t)b * Cc + g) * Dd + t];
    out[65536 + b * Dd + t] = ge * fin;                                    // guid_emb
    for (int idx = t; idx < 4 * Dd; idx += 128) {                          // aug
        int seg = idx >> 7, d = idx & 127;
        float v = (seg == 0) ? uc[b * Dd + d]
                : (seg == 1) ? cur[b * Dd + d]
                : (seg == 2) ? clu[((size_t)b * Cc + g) * Dd + d]
                             : depot[b * Dd + d];
        out[b * 4 * Dd + idx] = v * fin;
    }
}

extern "C" void kernel_launch(void* const* d_in, const int* in_sizes, int n_in,
                              void* d_out, int out_size, void* d_ws, size_t ws_size,
                              hipStream_t stream) {
    (void)in_sizes; (void)n_in; (void)out_size; (void)ws_size;
    const float* depot = (const float*)d_in[0];
    const float* clu   = (const float*)d_in[1];
    const float* cur   = (const float*)d_in[2];
    const float* node  = (const float*)d_in[3];
    const int*   isnew = (const int*)d_in[4];
    const int*   cmask = (const int*)d_in[5];
    const int*   visited = (const int*)d_in[6];
    const int*   mask  = (const int*)d_in[7];
    const float* Wq  = (const float*)d_in[8];
    const float* Wk  = (const float*)d_in[9];
    const float* Wv  = (const float*)d_in[10];
    const float* Wks = (const float*)d_in[11];
    const float* Wmq = (const float*)d_in[12];
    const float* Wmk = (const float*)d_in[13];
    const float* Wmv = (const float*)d_in[14];
    const float* Wmo = (const float*)d_in[15];

    char* ws = (char*)d_ws;
    size_t off = 0;
    auto carve = [&](size_t bytes) { char* p = ws + off; off = (off + bytes + 255) & ~(size_t)255; return p; };
    float*  um   = (float*) carve(Bsz * Dd * 4);                     // 64 KB
    float*  uc   = (float*) carve(Bsz * Dd * 4);
    float*  qh   = (float*) carve(Bsz * Dd * 4);
    float*  Wqq  = (float*) carve(3 * Dd * Dd * 4);                  // 192 KB
    float*  pum  = (float*) carve((size_t)Bsz * SEG * Dd * 4);       // 640 KB
    float*  puc  = (float*) carve((size_t)Bsz * SEG * Dd * 4);
    __bf16* WkkP = (__bf16*)carve(Dd * Dd * 2);                      // 32 KB each
    __bf16* WvvP = (__bf16*)carve(Dd * Dd * 2);
    __bf16* WksP = (__bf16*)carve(Dd * Dd * 2);
    __bf16* cluA = (__bf16*)carve((size_t)Bsz * MT * 4 * 512 * 2);   // 3.5 MB
    float*  kh   = (float*) carve((size_t)Bsz * MPAD * Dd * 4);      // 7 MB each
    float*  vh   = (float*) carve((size_t)Bsz * MPAD * Dd * 4);
    float*  pk   = (float*) carve((size_t)Bsz * MPAD * Dd * 4);

    k_packA<<<(Bsz * MT * 4 * 512) / 256, 256, 0, stream>>>(clu, cluA);
    k_packW<<<(3 * 16384) / 256, 256, 0, stream>>>(Wk, Wmk, Wv, Wmv, Wks, WkkP, WvvP, WksP);
    k_wqq  <<<(3 * Dd * Dd) / 256, 256, 0, stream>>>(Wq, Wmq, Wqq);
    k_means_part<<<dim3(SEG, Bsz), 256, 0, stream>>>(node, mask, cmask, pum, puc);
    k_means_reduce<<<Bsz, 128, 0, stream>>>(pum, puc, um, uc);
    k_gemm <<<dim3(MT, Bsz, 3), 256, 0, stream>>>(cluA, WkkP, WvvP, WksP, kh, vh, pk);
    k_qpath<<<Bsz, 128, 0, stream>>>(um, cur, depot, Wqq, qh);
    k_attn <<<Bsz, 128, 0, stream>>>(qh, kh, vh, pk, clu, uc, cur, depot, isnew, visited, Wmo, (float*)d_out);
}
